// BiRNNLayers_56547539419837
// MI455X (gfx1250) — matrix operand order repaired and checked by reading, not compiled, not run on hardware
//
#include <hip/hip_runtime.h>
#include <hip/hip_bf16.h>

// ---------------------------------------------------------------------------
// 2-layer BiLSTM text classifier for MI455X (gfx1250, wave32, WMMA).
// Mixed precision: f16 matmul operands, f32 accumulate (v_wmma_f32_16x16x32_f16),
// f32 cell state. Recurrence runs as 2 persistent workgroups (fwd/bwd) with Wh
// pre-packed into WMMA B-fragment layout in LDS, c/h state in VGPRs, and the
// input projection pre-stored in WMMA C-fragment layout so per-step loads are
// single b128s, double-buffered one timestep ahead.
// ---------------------------------------------------------------------------

typedef __attribute__((ext_vector_type(16))) _Float16 v16h;
typedef __attribute__((ext_vector_type(8)))  _Float16 v8h;
typedef __attribute__((ext_vector_type(8)))  float    v8f;

union V16H { v16h v; _Float16 h[16]; };
union V8H  { v8h  v; _Float16 h[8]; };
union V8F  { v8f  v; float     f[8]; };

#define BB 64
#define TT 512
#define HH 128
#define G4H 512   // 4*H gate width

__device__ __forceinline__ v8f wmma16(v16h a, v16h b, v8f c) {
  // D = A(16x32 f16) * B(32x16 f16) + C(16x16 f32)
  return __builtin_amdgcn_wmma_f32_16x16x32_f16(false, a, false, b, (short)0, c,
                                                false, false);
}

__device__ __forceinline__ float fast_rcp(float x) {
#if __has_builtin(__builtin_amdgcn_rcpf)
  return __builtin_amdgcn_rcpf(x);
#else
  return 1.0f / x;
#endif
}
__device__ __forceinline__ float fast_tanh(float x) {
#if __has_builtin(__builtin_amdgcn_tanhf)
  return __builtin_amdgcn_tanhf(x);
#else
  return tanhf(x);
#endif
}
__device__ __forceinline__ float sigf(float x) {
  return fast_rcp(1.0f + __expf(-x));
}

// ---------------------------------------------------------------------------
// Pack W (K x 512, f32 row-major) into WMMA B-fragment layout, f16:
//   P[((kt*32 + nt)*32 + lane)*16 + j] = W[kt*32 + (lane<16?0:8) + (j&7) + (j&8?16:0)][nt*16 + lane%16]
// so a lane's whole fragment is one contiguous 32B chunk.
// ---------------------------------------------------------------------------
__global__ void pack_w_kernel(const float* __restrict__ W, _Float16* __restrict__ P,
                              int total) {
  int idx = blockIdx.x * blockDim.x + threadIdx.x;
  if (idx >= total) return;
  int j    = idx & 15;
  int lane = (idx >> 4) & 31;
  int tile = idx >> 9;
  int nt = tile & 31;
  int kt = tile >> 5;
  int k = kt * 32 + ((lane < 16) ? 0 : 8) + (j & 7) + ((j & 8) ? 16 : 0);
  int n = nt * 16 + (lane & 15);
  P[idx] = (_Float16)W[k * 512 + n];
}

// ---------------------------------------------------------------------------
// Embedding gather: h0[b*T+t][e] = emb[x[b*T+t]][e]  (f32 -> f16)
// ---------------------------------------------------------------------------
__global__ void embed_kernel(const int* __restrict__ x, const float* __restrict__ emb,
                             _Float16* __restrict__ h0) {
  int idx = blockIdx.x * blockDim.x + threadIdx.x;
  if (idx >= BB * TT * HH) return;
  int e = idx & (HH - 1);
  int a = idx >> 7;
  h0[idx] = (_Float16)emb[(size_t)x[a] * HH + e];
}

// ---------------------------------------------------------------------------
// Projection GEMM: z_x = A(b*T+t, 0:K) @ W(K,512) + bias
// A: f16 row-major (M=32768 x K). WP: packed B-fragments.
// Output stored in the scan's C-fragment layout, f16:
//   xp[((((t*4 + rb)*32 + nt)*32 + lane)*8 + v]  where the 16x16 tile (rb,nt)
//   holds batch rows rb*16..+15 / gate cols nt*16..+15 and (lane,v) follow the
//   f32 C/D VGPR layout (M = v + (lane<16?0:8), N = lane%16).
// One block = 16 rows (= 16 consecutive t of one b) x 512 cols; 8 waves.
// ---------------------------------------------------------------------------
template <int K>
__global__ void __launch_bounds__(256)
proj_gemm_kernel(const _Float16* __restrict__ A, const _Float16* __restrict__ WP,
                 const float* __restrict__ bias, _Float16* __restrict__ xp) {
  __shared__ _Float16 sA[16 * K];  // 16 rows x K, <= 8 KB
  const int tid  = threadIdx.x;
  const int lane = tid & 31;
  const int wave = tid >> 5;
  const int hi8  = (lane < 16) ? 0 : 8;
  const int nlo  = lane & 15;
  const int row0 = blockIdx.x * 16;

  {
    const size_t abase = (size_t)row0 * K;
    for (int i = tid; i < 16 * K; i += 256) sA[i] = A[abase + i];
  }
  __syncthreads();

  V8F acc[4];
  const int n0 = wave * 64 + nlo;
#pragma unroll
  for (int q = 0; q < 4; ++q) {
    const float bv = bias[n0 + q * 16];
#pragma unroll
    for (int v = 0; v < 8; ++v) acc[q].f[v] = bv;
  }

#pragma unroll
  for (int kt = 0; kt < (K >> 5); ++kt) {
    // A fragment (16x32) from LDS: two contiguous 16B chunks per lane
    V16H af;
    const _Float16* pA = sA + nlo * K + kt * 32 + ((lane < 16) ? 0 : 8);
#pragma unroll
    for (int j = 0; j < 8; ++j) { af.h[j] = pA[j]; af.h[8 + j] = pA[16 + j]; }
#pragma unroll
    for (int q = 0; q < 4; ++q) {
      const int nt = wave * 4 + q;
      V16H bf;
      bf.v = *(const v16h*)&WP[(((size_t)kt * 32 + nt) * 32 + lane) * 16];
      acc[q].v = wmma16(af.v, bf.v, acc[q].v);
    }
  }

  // store into scan C-fragment layout: within this block b is constant,
  // only t varies across v.
  const int b    = row0 >> 9;
  const int rb   = b >> 4;
  const int mloc = b & 15;
  const int lane2 = nlo + ((mloc & 8) ? 16 : 0);
  const int v2    = mloc & 7;
#pragma unroll
  for (int q = 0; q < 4; ++q) {
    const int nt = wave * 4 + q;
#pragma unroll
    for (int v = 0; v < 8; ++v) {
      const int t = (row0 + hi8 + v) & (TT - 1);
      xp[((((size_t)t * 4 + rb) * 32 + nt) * 32 + lane2) * 8 + v2] =
          (_Float16)acc[q].f[v];
    }
  }
}

// ---------------------------------------------------------------------------
// Persistent LSTM scan. grid=2 (blockIdx.x: 0=forward, 1=backward), block=512
// (16 waves). Per step: z = xp_t + h_{t-1} @ Wh (WMMA, M=64 K=128 N=512),
// then gates. Wave w owns row-tile r=w&3 and gate-column groups
// cg in {2*(w>>2), 2*(w>>2)+1}: its 8 output tiles are the (i,f,g,o)
// quadruples for fixed (r,cg), so cell state never leaves VGPRs.
// xp fragments for step t+1 are prefetched into registers during step t.
// h_t is written back to LDS already in A-fragment layout.
// LDS: 128KB Wh + 16KB h + 4KB mask bits = 148KB.
// ---------------------------------------------------------------------------
__global__ void __launch_bounds__(512)
lstm_scan_kernel(const _Float16* __restrict__ xp_f, const _Float16* __restrict__ xp_b,
                 const _Float16* __restrict__ WhP_f, const _Float16* __restrict__ WhP_b,
                 const int* __restrict__ x, _Float16* __restrict__ out) {
  __shared__ _Float16 sWh[4 * 32 * 32 * 16];  // 128 KB: packed B fragments
  __shared__ _Float16 sA[16 * 32 * 16];       //  16 KB: h in A-fragment layout
  __shared__ unsigned smaskbits[BB * 16];     //   4 KB: (x != 0) bitmask

  const int dir = blockIdx.x;
  const bool rev = (dir == 1);
  const _Float16* xp  = rev ? xp_b : xp_f;
  const _Float16* WhP = rev ? WhP_b : WhP_f;
  const int colofs = dir * HH;

  const int tid  = threadIdx.x;
  const int lane = tid & 31;
  const int wave = tid >> 5;
  const int r    = wave & 3;
  const int cg0  = (wave >> 2) * 2;
  const int hi8  = (lane < 16) ? 0 : 8;
  const int nlo  = lane & 15;

  {  // stage packed Wh into LDS with 16B vector copies
    const int4* src = (const int4*)WhP;
    int4* dst = (int4*)sWh;
    for (int i = tid; i < (4 * 32 * 32 * 16) / 8; i += 512) dst[i] = src[i];
  }
  for (int i = tid; i < 16 * 32 * 16; i += 512) sA[i] = (_Float16)0.0f;
  for (int i = tid; i < BB * 16; i += 512) {  // build mask bits once
    const int m = i >> 4, w = i & 15;
    unsigned bits = 0u;
    for (int j = 0; j < 32; ++j)
      bits |= (x[m * TT + w * 32 + j] != 0 ? 1u : 0u) << j;
    smaskbits[i] = bits;
  }
  __syncthreads();

  float c_state[2][8], h_state[2][8];
#pragma unroll
  for (int cc = 0; cc < 2; ++cc)
#pragma unroll
    for (int v = 0; v < 8; ++v) { c_state[cc][v] = 0.0f; h_state[cc][v] = 0.0f; }

  // xp fragment pointer for (t, combo cc, gate g): one contiguous 16B per lane
  auto xp_frag = [&](int t, int cc, int g) -> const v8h* {
    const int nt = g * 8 + (cg0 + cc);
    return (const v8h*)&xp[((((size_t)t * 4 + r) * 32 + nt) * 32 + lane) * 8];
  };

  V8H fcur[2][4], fnxt[2][4];
  {
    const int t0 = rev ? (TT - 1) : 0;
#pragma unroll
    for (int cc = 0; cc < 2; ++cc)
#pragma unroll
      for (int g = 0; g < 4; ++g) fcur[cc][g].v = *xp_frag(t0, cc, g);
  }

  for (int s = 0; s < TT; ++s) {
    const int t = rev ? (TT - 1 - s) : s;
    const int tn = rev ? (t > 0 ? t - 1 : 0) : (t < TT - 1 ? t + 1 : TT - 1);
    __syncthreads();  // prev-step sA writes visible
    V16H a[4];
#pragma unroll
    for (int kt = 0; kt < 4; ++kt)
      a[kt].v = *(const v16h*)&sA[((r * 4 + kt) * 32 + lane) * 16];
    __syncthreads();  // all waves consumed h_{t-1}; safe to overwrite sA below

    // prefetch next step's xp fragments; latency hides under WMMA + gates
#pragma unroll
    for (int cc = 0; cc < 2; ++cc)
#pragma unroll
      for (int g = 0; g < 4; ++g) fnxt[cc][g].v = *xp_frag(tn, cc, g);

#pragma unroll
    for (int cc = 0; cc < 2; ++cc) {
      const int cg = cg0 + cc;
      V8F acc[4];
      // z init = xp_t from prefetched fragments
#pragma unroll
      for (int g = 0; g < 4; ++g)
#pragma unroll
        for (int v = 0; v < 8; ++v) acc[g].f[v] = (float)fcur[cc][g].h[v];
      // z += h_{t-1} @ Wh : 16 WMMAs per combo
#pragma unroll
      for (int kt = 0; kt < 4; ++kt) {
#pragma unroll
        for (int g = 0; g < 4; ++g) {
          const int nt = g * 8 + cg;
          V16H bf;
          bf.v = *(const v16h*)&sWh[((kt * 32 + nt) * 32 + lane) * 16];
          acc[g].v = wmma16(a[kt].v, bf.v, acc[g].v);
        }
      }
      // gates + state update (f32), write h_t in A-fragment layout + to global
      const int col = cg * 16 + nlo;       // hidden index 0..127
      const int kp  = col & 31;
      const int kt2 = col >> 5;
      const int lanep_hi = (kp & 8) ? 16 : 0;
      const int jp = (kp & 7) + ((kp & 16) ? 8 : 0);
      const int tw = t >> 5, tb = t & 31;
#pragma unroll
      for (int v = 0; v < 8; ++v) {
        const int m = r * 16 + hi8 + v;    // batch row
        const float ii = sigf(acc[0].f[v]);
        const float ff = sigf(acc[1].f[v]);
        const float gg = fast_tanh(acc[2].f[v]);
        const float oo = sigf(acc[3].f[v]);
        const float cprev = c_state[cc][v];
        const float hprev = h_state[cc][v];
        float cn = ff * cprev + ii * gg;
        float hn = oo * fast_tanh(cn);
        const bool mk = (smaskbits[m * 16 + tw] >> tb) & 1u;
        cn = mk ? cn : cprev;
        hn = mk ? hn : hprev;
        c_state[cc][v] = cn;
        h_state[cc][v] = hn;
        const int lanep = (m & 15) + lanep_hi;
        sA[((r * 4 + kt2) * 32 + lanep) * 16 + jp] = (_Float16)hn;
        out[((size_t)m * TT + t) * (2 * HH) + colofs + col] = (_Float16)hn;
      }
    }
#pragma unroll
    for (int cc = 0; cc < 2; ++cc)
#pragma unroll
      for (int g = 0; g < 4; ++g) fcur[cc][g] = fnxt[cc][g];
  }
}

// ---------------------------------------------------------------------------
// Channel max + mean pool: h2 (B*T, 256) f16 -> feat (B, 1024) f32
// feat[b][t] = max_c h2, feat[b][512+t] = mean_c h2
// ---------------------------------------------------------------------------
__global__ void pool_kernel(const _Float16* __restrict__ h2, float* __restrict__ feat) {
  int row = blockIdx.x * blockDim.x + threadIdx.x;
  if (row >= BB * TT) return;
  const v8h* p = (const v8h*)(h2 + (size_t)row * (2 * HH));
  float mx = -3.4e38f, sm = 0.0f;
  for (int jc = 0; jc < (2 * HH) / 8; ++jc) {
    V8H c; c.v = p[jc];
#pragma unroll
    for (int j = 0; j < 8; ++j) {
      float v = (float)c.h[j];
      mx = fmaxf(mx, v);
      sm += v;
    }
  }
  int b = row >> 9, t = row & (TT - 1);
  feat[b * 1024 + t] = mx;
  feat[b * 1024 + 512 + t] = sm * (1.0f / 256.0f);
}

// feat (64,1024) @ fc_W (1024,10) + fc_b, relu -> out (64,10) f32
__global__ void fc_kernel(const float* __restrict__ feat, const float* __restrict__ W,
                          const float* __restrict__ bias, float* __restrict__ out) {
  int idx = blockIdx.x * blockDim.x + threadIdx.x;
  if (idx >= BB * 10) return;
  int b = idx / 10, cls = idx % 10;
  float s = bias[cls];
  const float* f = feat + b * 1024;
  for (int k = 0; k < 1024; ++k) s += f[k] * W[k * 10 + cls];
  out[idx] = fmaxf(s, 0.0f);
}

// ---------------------------------------------------------------------------
extern "C" void kernel_launch(void* const* d_in, const int* in_sizes, int n_in,
                              void* d_out, int out_size, void* d_ws, size_t ws_size,
                              hipStream_t stream) {
  (void)in_sizes; (void)n_in; (void)out_size; (void)ws_size;
  const int*   x     = (const int*)  d_in[0];
  const float* emb   = (const float*)d_in[1];
  const float* Wx_f0 = (const float*)d_in[2];
  const float* Wh_f0 = (const float*)d_in[3];
  const float* b_f0  = (const float*)d_in[4];
  const float* Wx_b0 = (const float*)d_in[5];
  const float* Wh_b0 = (const float*)d_in[6];
  const float* b_b0  = (const float*)d_in[7];
  const float* Wx_f1 = (const float*)d_in[8];
  const float* Wh_f1 = (const float*)d_in[9];
  const float* b_f1  = (const float*)d_in[10];
  const float* Wx_b1 = (const float*)d_in[11];
  const float* Wh_b1 = (const float*)d_in[12];
  const float* b_b1  = (const float*)d_in[13];
  const float* fc_W  = (const float*)d_in[14];
  const float* fc_b  = (const float*)d_in[15];
  float* out = (float*)d_out;

  char* ws = (char*)d_ws;
  size_t off = 0;
  auto carve = [&](size_t bytes) -> char* {
    char* p = ws + off;
    off += (bytes + 255) & ~(size_t)255;
    return p;
  };

  _Float16* h0   = (_Float16*)carve((size_t)BB * TT * HH * 2);        // 8 MB
  _Float16* h1in = (_Float16*)carve((size_t)BB * TT * 2 * HH * 2);    // 16 MB
  _Float16* h2   = (_Float16*)carve((size_t)BB * TT * 2 * HH * 2);    // 16 MB
  _Float16* xp_f = (_Float16*)carve((size_t)TT * BB * G4H * 2);       // 32 MB
  _Float16* xp_b = (_Float16*)carve((size_t)TT * BB * G4H * 2);       // 32 MB
  _Float16* WxP_f0 = (_Float16*)carve((size_t)128 * 512 * 2);
  _Float16* WhP_f0 = (_Float16*)carve((size_t)128 * 512 * 2);
  _Float16* WxP_b0 = (_Float16*)carve((size_t)128 * 512 * 2);
  _Float16* WhP_b0 = (_Float16*)carve((size_t)128 * 512 * 2);
  _Float16* WxP_f1 = (_Float16*)carve((size_t)256 * 512 * 2);
  _Float16* WhP_f1 = (_Float16*)carve((size_t)128 * 512 * 2);
  _Float16* WxP_b1 = (_Float16*)carve((size_t)256 * 512 * 2);
  _Float16* WhP_b1 = (_Float16*)carve((size_t)128 * 512 * 2);
  float* feat = (float*)carve((size_t)BB * 1024 * 4);

  // 1) pack weights f32 -> f16 WMMA B-fragment layout
  const int g128 = (128 * 512 + 255) / 256, g256 = (256 * 512 + 255) / 256;
  pack_w_kernel<<<g128, 256, 0, stream>>>(Wx_f0, WxP_f0, 128 * 512);
  pack_w_kernel<<<g128, 256, 0, stream>>>(Wh_f0, WhP_f0, 128 * 512);
  pack_w_kernel<<<g128, 256, 0, stream>>>(Wx_b0, WxP_b0, 128 * 512);
  pack_w_kernel<<<g128, 256, 0, stream>>>(Wh_b0, WhP_b0, 128 * 512);
  pack_w_kernel<<<g256, 256, 0, stream>>>(Wx_f1, WxP_f1, 256 * 512);
  pack_w_kernel<<<g128, 256, 0, stream>>>(Wh_f1, WhP_f1, 128 * 512);
  pack_w_kernel<<<g256, 256, 0, stream>>>(Wx_b1, WxP_b1, 256 * 512);
  pack_w_kernel<<<g128, 256, 0, stream>>>(Wh_b1, WhP_b1, 128 * 512);

  // 2) embedding gather
  embed_kernel<<<(BB * TT * HH + 255) / 256, 256, 0, stream>>>(x, emb, h0);

  // 3) layer 0: input projections (parallel WMMA GEMMs), then scan
  const int gM = (BB * TT) / 16;  // 2048
  proj_gemm_kernel<128><<<gM, 256, 0, stream>>>(h0, WxP_f0, b_f0, xp_f);
  proj_gemm_kernel<128><<<gM, 256, 0, stream>>>(h0, WxP_b0, b_b0, xp_b);
  lstm_scan_kernel<<<2, 512, 0, stream>>>(xp_f, xp_b, WhP_f0, WhP_b0, x, h1in);

  // 4) layer 1 (input is concat(fwd,bwd) = h1in, K=256)
  proj_gemm_kernel<256><<<gM, 256, 0, stream>>>(h1in, WxP_f1, b_f1, xp_f);
  proj_gemm_kernel<256><<<gM, 256, 0, stream>>>(h1in, WxP_b1, b_b1, xp_b);
  lstm_scan_kernel<<<2, 512, 0, stream>>>(xp_f, xp_b, WhP_f1, WhP_b1, x, h2);

  // 5) pool + classifier head
  pool_kernel<<<(BB * TT + 255) / 256, 256, 0, stream>>>(h2, feat);
  fc_kernel<<<(BB * 10 + 63) / 64, 64, 0, stream>>>(feat, fc_W, fc_b, out);
}
